// DMS_STAttention_87205015978733
// MI455X (gfx1250) — matrix operand order repaired and checked by reading, not compiled, not run on hardware
//
#include <hip/hip_runtime.h>
#include <hip/hip_bf16.h>

typedef __attribute__((ext_vector_type(16))) _Float16 v16h;
typedef __attribute__((ext_vector_type(8)))  float    v8f;

#define WMMA_F16(a, b, c) \
  __builtin_amdgcn_wmma_f32_16x16x32_f16(false, (a), false, (b), (short)0, (c), false, false)

// ---------------------------------------------------------------------------
// WMMA fragment loaders, wave32 layouts per CDNA5 ISA §7.12.2
// A (16x32 f16, MxK): lanes 0-15 -> K 0..7 (v0..3) & 16..23 (v4..7); lanes 16-31 -> +8
// B (32x16 f16, KxN): lane = N col; lanes 0-15 -> K 0..7 & 16..23; lanes 16-31 -> +8 (same as A)
// C/D (16x16 f32):    lane&15 = N; vgpr r -> M = r + 8*(lane>=16)
// ---------------------------------------------------------------------------
__device__ __forceinline__ v16h frag_a(const _Float16* A, int lda) {
  int lane = threadIdx.x & 31;
  int m = lane & 15, hf = lane >> 4;
  v16h f;
#pragma unroll
  for (int v = 0; v < 8; ++v) {
    int kb = (v >> 2) * 16 + (v & 3) * 2 + hf * 8;
    f[2 * v]     = A[m * lda + kb];
    f[2 * v + 1] = A[m * lda + kb + 1];
  }
  return f;
}

__device__ __forceinline__ v16h frag_b(const _Float16* B, int ldb) {
  int lane = threadIdx.x & 31;
  int n = lane & 15, hf = lane >> 4;
  v16h f;
#pragma unroll
  for (int v = 0; v < 8; ++v) {
    int k = (v >> 2) * 16 + (v & 3) * 2 + hf * 8;
    f[2 * v]     = B[k * ldb + n];
    f[2 * v + 1] = B[(k + 1) * ldb + n];
  }
  return f;
}

// ---------------------------------------------------------------------------
// Precompute v_proj[l][c][p]: p<4 -> src head p, p>=4 -> dst head p-4.
// v[c,h] = sum_{cc<16} a[h,cc] * W[h*16+cc, c]
// ---------------------------------------------------------------------------
__global__ void projv_kernel(const float* __restrict__ lin,
                             const float* __restrict__ asrc,
                             const float* __restrict__ adst,
                             float* __restrict__ vproj) {
  int idx = blockIdx.x * blockDim.x + threadIdx.x;
  if (idx >= 6 * 64 * 8) return;
  int p = idx & 7, c = (idx >> 3) & 63, l = idx >> 9;
  int h = p & 3;
  const float* av = (p < 4 ? asrc : adst) + (l * 4 + h) * 16;
  const float* W  = lin + l * 64 * 64;
  float acc = 0.f;
#pragma unroll
  for (int cc = 0; cc < 16; ++cc) acc += av[cc] * W[(h * 16 + cc) * 64 + c];
  vproj[idx] = acc;
}

// ---------------------------------------------------------------------------
// GAT scores: sout[g][n][p] = sum_c x[g,n,c] * vproj[c,p]   (p in 0..7)
// Generic gather: addr = base(g) + n*SN + c*SC, base = (g/GDIV)*S0 + (g%GDIV)*S1
// ---------------------------------------------------------------------------
__global__ void scores_kernel(const float* __restrict__ xsrc,
                              long long S0, long long S1, int GDIV,
                              int SN, int SC, int N,
                              const float* __restrict__ vproj,
                              float* __restrict__ sout, int G) {
  int idx = blockIdx.x * blockDim.x + threadIdx.x;
  int total = G * N * 8;
  if (idx >= total) return;
  int p = idx & 7;
  int n = (idx >> 3) % N;
  int g = idx / (8 * N);
  const float* xb = xsrc + (long long)(g / GDIV) * S0 + (long long)(g % GDIV) * S1
                         + (long long)n * SN;
  float acc = 0.f;
#pragma unroll 8
  for (int c = 0; c < 64; ++c) acc += xb[(long long)c * SC] * vproj[c * 8 + p];
  sout[idx] = acc;
}

// ---------------------------------------------------------------------------
// Diff-pool: x_out(K,64) = softmax(relu(x@W1^T + b1)@W2^T + b2)^T @ x
// One wave per graph; all three GEMMs via v_wmma_f32_16x16x32_f16.
// ---------------------------------------------------------------------------
template <int N, int K>
__global__ void pool_kernel(const float* __restrict__ xsrc,
                            long long S0, long long S1, int GDIV, int SN, int SC,
                            const float* __restrict__ W1, const float* __restrict__ b1,
                            const float* __restrict__ W2, const float* __restrict__ b2,
                            float* __restrict__ xout) {
  constexpr int NP = (N + 31) & ~31;  // row pad (M tiles & reduction pad for S^T x)
  constexpr int KP = (K + 15) & ~15;  // assignment pad
  __shared__ _Float16 xh[NP * 64];
  __shared__ _Float16 w1t[64 * 64];
  __shared__ _Float16 th[NP * 64];
  __shared__ _Float16 w2t[64 * KP];
  __shared__ float    lg[NP * KP];
  __shared__ _Float16 st[KP * NP];

  int g = blockIdx.x;
  int tid = threadIdx.x;
  int hf = tid >> 4, ln = tid & 15;
  const float* xb = xsrc + (long long)(g / GDIV) * S0 + (long long)(g % GDIV) * S1;

  for (int idx = tid; idx < NP * 64; idx += 32) {
    int n = idx >> 6, c = idx & 63;
    float v = (n < N) ? xb[(long long)n * SN + (long long)c * SC] : 0.f;
    xh[idx] = (_Float16)v;
  }
  for (int idx = tid; idx < 64 * 64; idx += 32) {
    int c = idx >> 6, co = idx & 63;
    w1t[idx] = (_Float16)W1[co * 64 + c];             // W1^T
  }
  for (int idx = tid; idx < 64 * KP; idx += 32) {
    int c = idx / KP, ko = idx % KP;
    w2t[idx] = (ko < K) ? (_Float16)W2[ko * 64 + c] : (_Float16)0.f;
  }
  for (int idx = tid; idx < KP * NP; idx += 32) st[idx] = (_Float16)0.f;
  __syncthreads();

  // GEMM1: t = relu(x @ W1^T + b1), (NP x 64)
  for (int mt = 0; mt < NP / 16; ++mt)
    for (int nt = 0; nt < 4; ++nt) {
      v8f c8 = {0, 0, 0, 0, 0, 0, 0, 0};
#pragma unroll
      for (int ks = 0; ks < 2; ++ks) {
        v16h a = frag_a(&xh[(mt * 16) * 64 + ks * 32], 64);
        v16h b = frag_b(&w1t[(ks * 32) * 64 + nt * 16], 64);
        c8 = WMMA_F16(a, b, c8);
      }
#pragma unroll
      for (int r = 0; r < 8; ++r) {
        int m = mt * 16 + hf * 8 + r, n = nt * 16 + ln;
        float v = c8[r] + b1[n];
        th[m * 64 + n] = (_Float16)(v > 0.f ? v : 0.f);
      }
    }
  __syncthreads();

  // GEMM2: logits = t @ W2^T + b2, (NP x KP)
  for (int mt = 0; mt < NP / 16; ++mt)
    for (int nt = 0; nt < KP / 16; ++nt) {
      v8f c8 = {0, 0, 0, 0, 0, 0, 0, 0};
#pragma unroll
      for (int ks = 0; ks < 2; ++ks) {
        v16h a = frag_a(&th[(mt * 16) * 64 + ks * 32], 64);
        v16h b = frag_b(&w2t[(ks * 32) * KP + nt * 16], KP);
        c8 = WMMA_F16(a, b, c8);
      }
#pragma unroll
      for (int r = 0; r < 8; ++r) {
        int m = mt * 16 + hf * 8 + r, n = nt * 16 + ln;
        if (n < K) lg[m * KP + n] = c8[r] + b2[n];
      }
    }
  __syncthreads();

  // row softmax -> S^T (KP x NP), zero-padded
  for (int n = tid; n < N; n += 32) {
    float mx = -1e30f;
    for (int k2 = 0; k2 < K; ++k2) mx = fmaxf(mx, lg[n * KP + k2]);
    float e[K];
    float sum = 0.f;
    for (int k2 = 0; k2 < K; ++k2) { e[k2] = __expf(lg[n * KP + k2] - mx); sum += e[k2]; }
    float inv = 1.f / sum;
    for (int k2 = 0; k2 < K; ++k2) st[k2 * NP + n] = (_Float16)(e[k2] * inv);
  }
  __syncthreads();

  // GEMM3: x_out = S^T @ x, (K x 64)
  for (int mt = 0; mt < KP / 16; ++mt)
    for (int nt = 0; nt < 4; ++nt) {
      v8f c8 = {0, 0, 0, 0, 0, 0, 0, 0};
#pragma unroll
      for (int ks = 0; ks < NP / 32; ++ks) {
        v16h a = frag_a(&st[(mt * 16) * NP + ks * 32], NP);
        v16h b = frag_b(&xh[(ks * 32) * 64 + nt * 16], 64);
        c8 = WMMA_F16(a, b, c8);
      }
#pragma unroll
      for (int r = 0; r < 8; ++r) {
        int m = mt * 16 + hf * 8 + r, n = nt * 16 + ln;
        if (m < K) xout[(long long)g * K * 64 + m * 64 + n] = c8[r];
      }
    }
}

// ---------------------------------------------------------------------------
// Fuse: out[g,h] = A0 + L0@A1@R0 + L1@A2@R1, one wave per (g,h).
// A0/A1/A2 = softmax_i(leakyrelu(a1[j] + a2[i])) built from 8-wide scores.
// ---------------------------------------------------------------------------
template <int N0, int K0, int K1>
__global__ void fuse_kernel(const float* __restrict__ s0, const float* __restrict__ s1,
                            const float* __restrict__ s2,
                            const float* __restrict__ L0, const float* __restrict__ R0,
                            const float* __restrict__ L1, const float* __restrict__ R1,
                            float* __restrict__ out) {
  constexpr int NP  = (N0 + 15) & ~15;
  constexpr int K0P = (K0 + 31) & ~31;
  constexpr int K1P = (K1 + 31) & ~31;
  __shared__ float a1s[N0], a2s[N0], cmax[N0], csum[N0];
  __shared__ _Float16 A1[K0P * K0P], A2[K1P * K1P];
  __shared__ _Float16 L0h[NP * K0P], R0h[K0P * NP];
  __shared__ _Float16 L1h[NP * K1P], R1h[K1P * NP];
  __shared__ _Float16 T1[NP * K0P], T2[NP * K1P];

  int g = blockIdx.x >> 2, h = blockIdx.x & 3;
  int tid = threadIdx.x;
  int hf = tid >> 4, ln = tid & 15;

  for (int i = tid; i < N0; i += 32) {
    a1s[i] = s0[((long long)g * N0 + i) * 8 + h];
    a2s[i] = s0[((long long)g * N0 + i) * 8 + 4 + h];
  }
  for (int idx = tid; idx < K0P * K0P; idx += 32) A1[idx] = (_Float16)0.f;
  for (int idx = tid; idx < K1P * K1P; idx += 32) A2[idx] = (_Float16)0.f;
  for (int idx = tid; idx < NP * K0P; idx += 32) {
    int i = idx / K0P, k = idx % K0P;
    L0h[idx] = (i < N0 && k < K0) ? (_Float16)L0[(h * N0 + i) * K0 + k] : (_Float16)0.f;
  }
  for (int idx = tid; idx < K0P * NP; idx += 32) {
    int k = idx / NP, j = idx % NP;
    R0h[idx] = (k < K0 && j < N0) ? (_Float16)R0[(h * K0 + k) * N0 + j] : (_Float16)0.f;
  }
  for (int idx = tid; idx < NP * K1P; idx += 32) {
    int i = idx / K1P, k = idx % K1P;
    L1h[idx] = (i < N0 && k < K1) ? (_Float16)L1[(h * N0 + i) * K1 + k] : (_Float16)0.f;
  }
  for (int idx = tid; idx < K1P * NP; idx += 32) {
    int k = idx / NP, j = idx % NP;
    R1h[idx] = (k < K1 && j < N0) ? (_Float16)R1[(h * K1 + k) * N0 + j] : (_Float16)0.f;
  }
  __syncthreads();

  // A0 column statistics (softmax over i for fixed j)
  for (int j = tid; j < N0; j += 32) {
    float bj = a1s[j];
    float mx = -1e30f;
    for (int i = 0; i < N0; ++i) {
      float e = bj + a2s[i];
      e = (e >= 0.f) ? e : 0.2f * e;
      mx = fmaxf(mx, e);
    }
    float s = 0.f;
    for (int i = 0; i < N0; ++i) {
      float e = bj + a2s[i];
      e = (e >= 0.f) ? e : 0.2f * e;
      s += __expf(e - mx);
    }
    cmax[j] = mx; csum[j] = s;
  }
  // A1 softmax (K0 x K0)
  for (int j = tid; j < K0; j += 32) {
    float bj = s1[((long long)g * K0 + j) * 8 + h];
    float mx = -1e30f;
    for (int i = 0; i < K0; ++i) {
      float e = bj + s1[((long long)g * K0 + i) * 8 + 4 + h];
      e = (e >= 0.f) ? e : 0.2f * e;
      mx = fmaxf(mx, e);
    }
    float s = 0.f;
    for (int i = 0; i < K0; ++i) {
      float e = bj + s1[((long long)g * K0 + i) * 8 + 4 + h];
      e = (e >= 0.f) ? e : 0.2f * e;
      s += __expf(e - mx);
    }
    float inv = 1.f / s;
    for (int i = 0; i < K0; ++i) {
      float e = bj + s1[((long long)g * K0 + i) * 8 + 4 + h];
      e = (e >= 0.f) ? e : 0.2f * e;
      A1[i * K0P + j] = (_Float16)(__expf(e - mx) * inv);
    }
  }
  // A2 softmax (K1 x K1)
  for (int j = tid; j < K1; j += 32) {
    float bj = s2[((long long)g * K1 + j) * 8 + h];
    float mx = -1e30f;
    for (int i = 0; i < K1; ++i) {
      float e = bj + s2[((long long)g * K1 + i) * 8 + 4 + h];
      e = (e >= 0.f) ? e : 0.2f * e;
      mx = fmaxf(mx, e);
    }
    float s = 0.f;
    for (int i = 0; i < K1; ++i) {
      float e = bj + s2[((long long)g * K1 + i) * 8 + 4 + h];
      e = (e >= 0.f) ? e : 0.2f * e;
      s += __expf(e - mx);
    }
    float inv = 1.f / s;
    for (int i = 0; i < K1; ++i) {
      float e = bj + s2[((long long)g * K1 + i) * 8 + 4 + h];
      e = (e >= 0.f) ? e : 0.2f * e;
      A2[i * K1P + j] = (_Float16)(__expf(e - mx) * inv);
    }
  }
  __syncthreads();

  // T1 = L0 @ A1 ; T2 = L1 @ A2
  for (int mt = 0; mt < NP / 16; ++mt)
    for (int nt = 0; nt < K0P / 16; ++nt) {
      v8f c8 = {0, 0, 0, 0, 0, 0, 0, 0};
#pragma unroll
      for (int ks = 0; ks < K0P / 32; ++ks) {
        v16h a = frag_a(&L0h[(mt * 16) * K0P + ks * 32], K0P);
        v16h b = frag_b(&A1[(ks * 32) * K0P + nt * 16], K0P);
        c8 = WMMA_F16(a, b, c8);
      }
#pragma unroll
      for (int r = 0; r < 8; ++r)
        T1[(mt * 16 + hf * 8 + r) * K0P + nt * 16 + ln] = (_Float16)c8[r];
    }
  for (int mt = 0; mt < NP / 16; ++mt)
    for (int nt = 0; nt < K1P / 16; ++nt) {
      v8f c8 = {0, 0, 0, 0, 0, 0, 0, 0};
#pragma unroll
      for (int ks = 0; ks < K1P / 32; ++ks) {
        v16h a = frag_a(&L1h[(mt * 16) * K1P + ks * 32], K1P);
        v16h b = frag_b(&A2[(ks * 32) * K1P + nt * 16], K1P);
        c8 = WMMA_F16(a, b, c8);
      }
#pragma unroll
      for (int r = 0; r < 8; ++r)
        T2[(mt * 16 + hf * 8 + r) * K1P + nt * 16 + ln] = (_Float16)c8[r];
    }
  __syncthreads();

  // out = A0 + T1@R0 + T2@R1
  long long obase = (long long)(g * 4 + h) * N0 * N0;
  for (int mt = 0; mt < NP / 16; ++mt)
    for (int nt = 0; nt < NP / 16; ++nt) {
      v8f c8 = {0, 0, 0, 0, 0, 0, 0, 0};
#pragma unroll
      for (int ks = 0; ks < K0P / 32; ++ks) {
        v16h a = frag_a(&T1[(mt * 16) * K0P + ks * 32], K0P);
        v16h b = frag_b(&R0h[(ks * 32) * NP + nt * 16], NP);
        c8 = WMMA_F16(a, b, c8);
      }
#pragma unroll
      for (int ks = 0; ks < K1P / 32; ++ks) {
        v16h a = frag_a(&T2[(mt * 16) * K1P + ks * 32], K1P);
        v16h b = frag_b(&R1h[(ks * 32) * NP + nt * 16], NP);
        c8 = WMMA_F16(a, b, c8);
      }
#pragma unroll
      for (int r = 0; r < 8; ++r) {
        int i = mt * 16 + hf * 8 + r, j = nt * 16 + ln;
        if (i < N0 && j < N0) {
          float e = a1s[j] + a2s[i];
          e = (e >= 0.f) ? e : 0.2f * e;
          float a0 = __expf(e - cmax[j]) / csum[j];
          out[obase + (long long)i * N0 + j] = a0 + c8[r];
        }
      }
    }
}

// ---------------------------------------------------------------------------
extern "C" void kernel_launch(void* const* d_in, const int* in_sizes, int n_in,
                              void* d_out, int out_size, void* d_ws, size_t ws_size,
                              hipStream_t stream) {
  (void)in_sizes; (void)n_in; (void)out_size; (void)ws_size;
  const float* src  = (const float*)d_in[0];
  const float* lin  = (const float*)d_in[1];
  const float* asrc = (const float*)d_in[2];
  const float* adst = (const float*)d_in[3];
  const float* sp0W1 = (const float*)d_in[4];  const float* sp0b1 = (const float*)d_in[5];
  const float* sp0W2 = (const float*)d_in[6];  const float* sp0b2 = (const float*)d_in[7];
  const float* sp1W1 = (const float*)d_in[8];  const float* sp1b1 = (const float*)d_in[9];
  const float* sp1W2 = (const float*)d_in[10]; const float* sp1b2 = (const float*)d_in[11];
  const float* tp0W1 = (const float*)d_in[12]; const float* tp0b1 = (const float*)d_in[13];
  const float* tp0W2 = (const float*)d_in[14]; const float* tp0b2 = (const float*)d_in[15];
  const float* tp1W1 = (const float*)d_in[16]; const float* tp1b1 = (const float*)d_in[17];
  const float* tp1W2 = (const float*)d_in[18]; const float* tp1b2 = (const float*)d_in[19];
  const float* sl0 = (const float*)d_in[20]; const float* sr0 = (const float*)d_in[21];
  const float* sl1 = (const float*)d_in[22]; const float* sr1 = (const float*)d_in[23];
  const float* tl0 = (const float*)d_in[24]; const float* tr0 = (const float*)d_in[25];
  const float* tl1 = (const float*)d_in[26]; const float* tr1 = (const float*)d_in[27];
  float* out = (float*)d_out;
  float* ws  = (float*)d_ws;

  const int GS = 4096, GT = 1600;
  // ws layout (floats)
  float* vproj = ws;                         // 6*64*8 = 3072
  float* s0s = vproj + 3072;                 // GS*25*8
  float* x1s = s0s + (long long)GS * 25 * 8; // GS*12*64
  float* s1s = x1s + (long long)GS * 12 * 64;
  float* x2s = s1s + (long long)GS * 12 * 8; // GS*5*64
  float* s2s = x2s + (long long)GS * 5 * 64; // GS*5*8
  float* s0t = s2s + (long long)GS * 5 * 8;  // GT*64*8
  float* x1t = s0t + (long long)GT * 64 * 8; // GT*32*64
  float* s1t = x1t + (long long)GT * 32 * 64;
  float* x2t = s1t + (long long)GT * 32 * 8; // GT*12*64
  float* s2t = x2t + (long long)GT * 12 * 64;// GT*12*8

  // src strides: src[b,c,t,j] flat = b*102400 + c*1600 + t*25 + j
  // spatial graph g=b*64+t: base=(g/64)*102400+(g%64)*25, n=j (SN=1), c (SC=1600)
  // temporal graph g=b*25+j: base=(g/25)*102400+(g%25)*1, n=t (SN=25), c (SC=1600)

  projv_kernel<<<12, 256, 0, stream>>>(lin, asrc, adst, vproj);

  // ---- spatial branch ----
  scores_kernel<<<(GS * 25 * 8 + 255) / 256, 256, 0, stream>>>(
      src, 102400, 25, 64, 1, 1600, 25, vproj + 0 * 512, s0s, GS);
  pool_kernel<25, 12><<<GS, 32, 0, stream>>>(
      src, 102400, 25, 64, 1, 1600, sp0W1, sp0b1, sp0W2, sp0b2, x1s);
  scores_kernel<<<(GS * 12 * 8 + 255) / 256, 256, 0, stream>>>(
      x1s, 12 * 64, 0, 1, 64, 1, 12, vproj + 1 * 512, s1s, GS);
  pool_kernel<12, 5><<<GS, 32, 0, stream>>>(
      x1s, 12 * 64, 0, 1, 64, 1, sp1W1, sp1b1, sp1W2, sp1b2, x2s);
  scores_kernel<<<(GS * 5 * 8 + 255) / 256, 256, 0, stream>>>(
      x2s, 5 * 64, 0, 1, 64, 1, 5, vproj + 2 * 512, s2s, GS);
  fuse_kernel<25, 12, 5><<<GS * 4, 32, 0, stream>>>(
      s0s, s1s, s2s, sl0, sr0, sl1, sr1, out);

  // ---- temporal branch ----
  scores_kernel<<<(GT * 64 * 8 + 255) / 256, 256, 0, stream>>>(
      src, 102400, 1, 25, 25, 1600, 64, vproj + 3 * 512, s0t, GT);
  pool_kernel<64, 32><<<GT, 32, 0, stream>>>(
      src, 102400, 1, 25, 25, 1600, tp0W1, tp0b1, tp0W2, tp0b2, x1t);
  scores_kernel<<<(GT * 32 * 8 + 255) / 256, 256, 0, stream>>>(
      x1t, 32 * 64, 0, 1, 64, 1, 32, vproj + 4 * 512, s1t, GT);
  pool_kernel<32, 12><<<GT, 32, 0, stream>>>(
      x1t, 32 * 64, 0, 1, 64, 1, tp1W1, tp1b1, tp1W2, tp1b2, x2t);
  scores_kernel<<<(GT * 12 * 8 + 255) / 256, 256, 0, stream>>>(
      x2t, 12 * 64, 0, 1, 64, 1, 12, vproj + 5 * 512, s2t, GT);
  fuse_kernel<64, 32, 12><<<GT * 4, 32, 0, stream>>>(
      s0t, s1t, s2t, tl0, tr0, tl1, tr1, out + (long long)GS * 4 * 25 * 25);
}
